// ChineseDecoder_19739669692788
// MI455X (gfx1250) — matrix operand order repaired and checked by reading, not compiled, not run on hardware
//
#include <hip/hip_runtime.h>

// ---------------- problem constants (from reference) ----------------
#define VCAB  32000
#define EDIM  512
#define HDIM  1024
#define ENCD  2048          // 2*H bidirectional encoder dim
#define BSZ   32
#define SLEN  128
#define TSTEPS 64
#define X0DIM (EDIM + ENCD)          // 2560
#define FTDIM (HDIM + ENCD + EDIM)   // 3584

typedef __bf16 bf16_t;
typedef __attribute__((ext_vector_type(16))) __bf16 v16bf;
typedef __attribute__((ext_vector_type(8)))  __bf16 v8bf;
typedef __attribute__((ext_vector_type(8)))  float  v8f;

__device__ __forceinline__ float sigmf(float x) { return 1.0f / (1.0f + __expf(-x)); }

// ---------------- fp32 -> bf16 conversion ----------------
__global__ __launch_bounds__(256)
void cvt_f32_bf16(const float* __restrict__ s, bf16_t* __restrict__ d, size_t n) {
  size_t i = (size_t)blockIdx.x * 256 + threadIdx.x;
  if (i < n) d[i] = (bf16_t)s[i];
}

// ---------------- initial GRU state h0 = h1 = encoder_hidden ----------------
__global__ __launch_bounds__(256)
void init_state(const float* __restrict__ eh, float* h0f, float* h1f,
                bf16_t* h0b, bf16_t* h1b) {
  int i = blockIdx.x * 256 + threadIdx.x;
  if (i < BSZ * HDIM) {
    float x = eh[i];
    h0f[i] = x; h1f[i] = x; h0b[i] = (bf16_t)x; h1b[i] = (bf16_t)x;
  }
}

// ---------------- WMMA GEMM: C[m,n] = act( sum_k A[m,k]*W[n,k] + bias[n] ) ----
// A: bf16 [M,K] row-major.  W: bf16 [N,K] row-major (computes A @ W^T).
// Each wave owns one 16-col n-tile and TWO 16-row m-tiles: B fragment is
// loaded once per k-step and fed to two independent WMMA accumulator chains
// (halves B-side L2 traffic, removes the serial WMMA RAW stall).
// C fp32 at C[m*ldc + n]; optional bf16 mirror Cbf (row stride N).
// grid = (N/128, M/32); block = 256 (8 waves).
__global__ __launch_bounds__(256)
void wmma_gemm_nt(const bf16_t* __restrict__ A, const bf16_t* __restrict__ W,
                  const float* __restrict__ bias, float* __restrict__ C,
                  bf16_t* __restrict__ Cbf, int N, int K, long ldc, int act) {
  const int wave = threadIdx.x >> 5;
  const int lane = threadIdx.x & 31;
  const int lo = lane & 15;
  const int hi = lane >> 4;
  const int ntile = blockIdx.x * 8 + wave;
  const int mbase = blockIdx.y * 32;           // two 16-row tiles per wave
  if (ntile * 16 >= N) return;   // wave-uniform; EXEC stays all-ones for WMMA

  const bf16_t* arow0 = A + (size_t)(mbase + lo) * (size_t)K;
  const bf16_t* arow1 = arow0 + (size_t)16 * (size_t)K;
  const bf16_t* wrow  = W + (size_t)(ntile * 16 + lo) * (size_t)K;

  v8f acc0 = {};
  v8f acc1 = {};
#pragma unroll 2
  for (int k0 = 0; k0 < K; k0 += 32) {
    // B fragment (32x16): lane lo = column n (= W row n); 16 contiguous K per half
    v16bf bfrag = *(const v16bf*)(wrow + k0 + hi * 16);
    // A fragments (16x32): 16 contiguous K per half (K = 16*hi + s), matching B
    union { v16bf v; v8bf h[2]; } af0, af1;
    af0.h[0] = *(const v8bf*)(arow0 + k0 + hi * 16);
    af0.h[1] = *(const v8bf*)(arow0 + k0 + hi * 16 + 8);
    af1.h[0] = *(const v8bf*)(arow1 + k0 + hi * 16);
    af1.h[1] = *(const v8bf*)(arow1 + k0 + hi * 16 + 8);
    acc0 = __builtin_amdgcn_wmma_f32_16x16x32_bf16(
        false, af0.v, false, bfrag, (short)0, acc0, false, false);
    acc1 = __builtin_amdgcn_wmma_f32_16x16x32_bf16(
        false, af1.v, false, bfrag, (short)0, acc1, false, false);
  }

  const int n = ntile * 16 + lo;
  const float bv = bias ? bias[n] : 0.0f;
#pragma unroll
  for (int r = 0; r < 8; ++r) {
    int m0 = mbase + r + hi * 8;          // C layout: lanes16-31 hold M=r+8
    int m1 = m0 + 16;
    float x0 = acc0[r] + bv;
    float x1 = acc1[r] + bv;
    if (act == 1) { x0 = tanhf(x0); x1 = tanhf(x1); }
    C[(long)m0 * ldc + n] = x0;
    C[(long)m1 * ldc + n] = x1;
    if (Cbf) {
      Cbf[(size_t)m0 * (size_t)N + n] = (bf16_t)x0;
      Cbf[(size_t)m1 * (size_t)N + n] = (bf16_t)x1;
    }
  }
}

// ---------------- attention + context + embed + x0/feat assembly ------------
// one block per batch row b; 256 threads
__global__ __launch_bounds__(256)
void attn_step(const float* __restrict__ q, const float* __restrict__ enc_proj,
               const float* __restrict__ enc_out, const unsigned char* __restrict__ mask,
               const float* __restrict__ vvec, const int* __restrict__ tokens,
               const float* __restrict__ emb_w, float* __restrict__ attns_out,
               bf16_t* __restrict__ x0, bf16_t* __restrict__ feat, int t) {
  __shared__ float e_sh[SLEN];
  __shared__ float a_sh[SLEN];
  __shared__ float red[2];
  const int b = blockIdx.x;
  const int tid = threadIdx.x;

  // scores e[s] = v . tanh(enc_proj[b,s,:] + q[b,:])
  for (int s = tid; s < SLEN; s += 256) {
    const float* ep = enc_proj + ((size_t)(b * SLEN + s)) * HDIM;
    const float* qp = q + (size_t)b * HDIM;
    float acc = 0.0f;
    for (int h = 0; h < HDIM; ++h) acc += vvec[h] * tanhf(ep[h] + qp[h]);
    e_sh[s] = mask[b * SLEN + s] ? acc : -1e9f;
  }
  __syncthreads();
  if (tid == 0) {
    float mx = e_sh[0];
    for (int s = 1; s < SLEN; ++s) mx = fmaxf(mx, e_sh[s]);
    float sum = 0.0f;
    for (int s = 0; s < SLEN; ++s) sum += __expf(e_sh[s] - mx);
    red[0] = mx; red[1] = sum;
  }
  __syncthreads();
  for (int s = tid; s < SLEN; s += 256) {
    float a = __expf(e_sh[s] - red[0]) / red[1];
    a_sh[s] = a;
    attns_out[((long)b * TSTEPS + t) * SLEN + s] = a;
  }
  __syncthreads();
  // ctx[d] = sum_s attn[s] * enc_out[b,s,d]  -> bf16 into x0[512:] and feat[1024:3072]
  for (int d = tid; d < ENCD; d += 256) {
    const float* eo = enc_out + ((size_t)b * SLEN) * ENCD + d;
    float acc = 0.0f;
    for (int s = 0; s < SLEN; ++s) acc += a_sh[s] * eo[(size_t)s * ENCD];
    bf16_t c = (bf16_t)acc;
    x0[(size_t)b * X0DIM + EDIM + d] = c;
    feat[(size_t)b * FTDIM + HDIM + d] = c;
  }
  // embedding gather (teacher forcing token) -> x0[0:512], feat[3072:3584]
  const int tok = tokens[b * (TSTEPS + 1) + t];
  for (int i = tid; i < EDIM; i += 256) {
    bf16_t ev = (bf16_t)emb_w[(size_t)tok * EDIM + i];
    x0[(size_t)b * X0DIM + i] = ev;
    feat[(size_t)b * FTDIM + ENCD + HDIM + i] = ev;
  }
}

// ---------------- GRU elementwise combine -----------------------------------
__global__ __launch_bounds__(256)
void gru_elem(const float* __restrict__ gi, const float* __restrict__ gh,
              const float* __restrict__ h_in, float* __restrict__ h_out,
              bf16_t* __restrict__ h_out_bf, bf16_t* __restrict__ h_out_bf2,
              long stride2) {
  int idx = blockIdx.x * 256 + threadIdx.x;
  if (idx >= BSZ * HDIM) return;
  int b = idx / HDIM, j = idx % HDIM;
  const float* gib = gi + (size_t)b * 3 * HDIM;
  const float* ghb = gh + (size_t)b * 3 * HDIM;
  float r = sigmf(gib[j] + ghb[j]);
  float z = sigmf(gib[HDIM + j] + ghb[HDIM + j]);
  float n = tanhf(gib[2 * HDIM + j] + r * ghb[2 * HDIM + j]);
  float h = (1.0f - z) * n + z * h_in[idx];
  h_out[idx] = h;
  h_out_bf[idx] = (bf16_t)h;
  if (h_out_bf2) h_out_bf2[(size_t)b * stride2 + j] = (bf16_t)h;
}

// ---------------- argmax over vocab -----------------------------------------
__global__ __launch_bounds__(256)
void argmax_vocab(const float* __restrict__ logits, float* __restrict__ preds, int t) {
  __shared__ float sv[256];
  __shared__ int   si[256];
  const int b = blockIdx.x;
  const float* row = logits + ((long)b * TSTEPS + t) * (long)VCAB;
  float best = -3.4e38f; int bi = 0;
  for (int i = threadIdx.x; i < VCAB; i += 256) {
    float x = row[i];
    if (x > best || (x == best && i < bi)) { best = x; bi = i; }
  }
  sv[threadIdx.x] = best; si[threadIdx.x] = bi;
  __syncthreads();
  for (int off = 128; off > 0; off >>= 1) {
    if (threadIdx.x < off) {
      float ov = sv[threadIdx.x + off]; int oi = si[threadIdx.x + off];
      if (ov > sv[threadIdx.x] || (ov == sv[threadIdx.x] && oi < si[threadIdx.x])) {
        sv[threadIdx.x] = ov; si[threadIdx.x] = oi;
      }
    }
    __syncthreads();
  }
  if (threadIdx.x == 0) preds[b * TSTEPS + t] = (float)si[0];
}

// ---------------- host orchestration ----------------------------------------
extern "C" void kernel_launch(void* const* d_in, const int* in_sizes, int n_in,
                              void* d_out, int out_size, void* d_ws, size_t ws_size,
                              hipStream_t stream) {
  (void)in_sizes; (void)n_in; (void)out_size; (void)ws_size;
  const float* encoder_hidden = (const float*)d_in[0];
  const float* enc_output     = (const float*)d_in[1];
  const unsigned char* src_mask = (const unsigned char*)d_in[2];
  const int*   target_tokens  = (const int*)d_in[3];
  const float* emb_w = (const float*)d_in[4];
  const float* Wk = (const float*)d_in[5];  const float* bk = (const float*)d_in[6];
  const float* Wq = (const float*)d_in[7];  const float* bq = (const float*)d_in[8];
  const float* vv = (const float*)d_in[9];
  const float* Wih0 = (const float*)d_in[10]; const float* Whh0 = (const float*)d_in[11];
  const float* bih0 = (const float*)d_in[12]; const float* bhh0 = (const float*)d_in[13];
  const float* Wih1 = (const float*)d_in[14]; const float* Whh1 = (const float*)d_in[15];
  const float* bih1 = (const float*)d_in[16]; const float* bhh1 = (const float*)d_in[17];
  const float* Wb = (const float*)d_in[18];  const float* bb = (const float*)d_in[19];

  float* out_logits = (float*)d_out;                               // [B,T,V]
  float* out_preds  = out_logits + (size_t)BSZ * TSTEPS * VCAB;    // [B,T]
  float* out_attns  = out_preds + (size_t)BSZ * TSTEPS;            // [B,T,S]

  // workspace carve-out (bf16 weight mirrors ~94MB fit in 192MB L2 ->
  // the 64-step recurrent loop streams weights from L2, not HBM)
  char* ws = (char*)d_ws;
  size_t off = 0;
  auto carve = [&](size_t bytes) -> void* {
    void* p = ws + off;
    off = (off + bytes + 255) & ~(size_t)255;
    return p;
  };
  bf16_t* embw_b = (bf16_t*)carve((size_t)VCAB * EDIM * 2);
  bf16_t* Wk_b   = (bf16_t*)carve((size_t)HDIM * ENCD * 2);
  bf16_t* Wq_b   = (bf16_t*)carve((size_t)HDIM * HDIM * 2);
  bf16_t* Wih0_b = (bf16_t*)carve((size_t)3 * HDIM * X0DIM * 2);
  bf16_t* Whh0_b = (bf16_t*)carve((size_t)3 * HDIM * HDIM * 2);
  bf16_t* Wih1_b = (bf16_t*)carve((size_t)3 * HDIM * HDIM * 2);
  bf16_t* Whh1_b = (bf16_t*)carve((size_t)3 * HDIM * HDIM * 2);
  bf16_t* Wb_b   = (bf16_t*)carve((size_t)EDIM * FTDIM * 2);
  bf16_t* enc_b  = (bf16_t*)carve((size_t)BSZ * SLEN * ENCD * 2);
  float*  enc_proj = (float*)carve((size_t)BSZ * SLEN * HDIM * 4);
  float*  h0f = (float*)carve((size_t)BSZ * HDIM * 4);
  float*  h1f = (float*)carve((size_t)BSZ * HDIM * 4);
  bf16_t* h0b = (bf16_t*)carve((size_t)BSZ * HDIM * 2);
  bf16_t* h1b = (bf16_t*)carve((size_t)BSZ * HDIM * 2);
  float*  qf  = (float*)carve((size_t)BSZ * HDIM * 4);
  bf16_t* x0b = (bf16_t*)carve((size_t)BSZ * X0DIM * 2);
  bf16_t* ftb = (bf16_t*)carve((size_t)BSZ * FTDIM * 2);
  float*  gi  = (float*)carve((size_t)BSZ * 3 * HDIM * 4);
  float*  gh  = (float*)carve((size_t)BSZ * 3 * HDIM * 4);
  float*  bottf = (float*)carve((size_t)BSZ * EDIM * 4);
  bf16_t* bottb = (bf16_t*)carve((size_t)BSZ * EDIM * 2);

  auto cvt = [&](const float* s, bf16_t* d, size_t n) {
    cvt_f32_bf16<<<(unsigned)((n + 255) / 256), 256, 0, stream>>>(s, d, n);
  };
  // bf16 mirrors (rerun each call: deterministic, no cached state)
  cvt(emb_w, embw_b, (size_t)VCAB * EDIM);
  cvt(Wk, Wk_b, (size_t)HDIM * ENCD);
  cvt(Wq, Wq_b, (size_t)HDIM * HDIM);
  cvt(Wih0, Wih0_b, (size_t)3 * HDIM * X0DIM);
  cvt(Whh0, Whh0_b, (size_t)3 * HDIM * HDIM);
  cvt(Wih1, Wih1_b, (size_t)3 * HDIM * HDIM);
  cvt(Whh1, Whh1_b, (size_t)3 * HDIM * HDIM);
  cvt(Wb, Wb_b, (size_t)EDIM * FTDIM);
  cvt(enc_output, enc_b, (size_t)BSZ * SLEN * ENCD);
  init_state<<<(BSZ * HDIM + 255) / 256, 256, 0, stream>>>(encoder_hidden, h0f, h1f, h0b, h1b);

  // enc_proj = enc_output @ Wk^T + bk   [B*S=4096, H], K=ENC
  wmma_gemm_nt<<<dim3(HDIM / 128, (BSZ * SLEN) / 32), 256, 0, stream>>>(
      enc_b, Wk_b, bk, enc_proj, nullptr, HDIM, ENCD, HDIM, 0);

  for (int t = 0; t < TSTEPS; ++t) {
    // q = h1 @ Wq^T + bq
    wmma_gemm_nt<<<dim3(HDIM / 128, 1), 256, 0, stream>>>(
        h1b, Wq_b, bq, qf, nullptr, HDIM, HDIM, HDIM, 0);
    // attention, context, embedding; assembles x0 (bf16) and feat pieces
    attn_step<<<BSZ, 256, 0, stream>>>(qf, enc_proj, enc_output, src_mask, vv,
                                       target_tokens, emb_w, out_attns, x0b, ftb, t);
    // GRU layer 0
    wmma_gemm_nt<<<dim3(3 * HDIM / 128, 1), 256, 0, stream>>>(
        x0b, Wih0_b, bih0, gi, nullptr, 3 * HDIM, X0DIM, 3 * HDIM, 0);
    wmma_gemm_nt<<<dim3(3 * HDIM / 128, 1), 256, 0, stream>>>(
        h0b, Whh0_b, bhh0, gh, nullptr, 3 * HDIM, HDIM, 3 * HDIM, 0);
    gru_elem<<<(BSZ * HDIM + 255) / 256, 256, 0, stream>>>(gi, gh, h0f, h0f, h0b,
                                                           (bf16_t*)nullptr, 0);
    // GRU layer 1 (input = new h0, state = old h1); new h1 also lands in feat[0:H]
    wmma_gemm_nt<<<dim3(3 * HDIM / 128, 1), 256, 0, stream>>>(
        h0b, Wih1_b, bih1, gi, nullptr, 3 * HDIM, HDIM, 3 * HDIM, 0);
    wmma_gemm_nt<<<dim3(3 * HDIM / 128, 1), 256, 0, stream>>>(
        h1b, Whh1_b, bhh1, gh, nullptr, 3 * HDIM, HDIM, 3 * HDIM, 0);
    gru_elem<<<(BSZ * HDIM + 255) / 256, 256, 0, stream>>>(gi, gh, h1f, h1f, h1b,
                                                           ftb, (long)FTDIM);
    // bottleneck: tanh(feat @ Wb^T + bb), keep bf16 mirror for vocab GEMM
    wmma_gemm_nt<<<dim3(EDIM / 128, 1), 256, 0, stream>>>(
        ftb, Wb_b, bb, bottf, bottb, EDIM, FTDIM, EDIM, 1);
    // logits = bott @ emb_w^T, scattered directly into d_out[B,T,V] via ldc
    wmma_gemm_nt<<<dim3(VCAB / 128, 1), 256, 0, stream>>>(
        bottb, embw_b, nullptr, out_logits + (long)t * VCAB, nullptr,
        VCAB, EDIM, (long)TSTEPS * VCAB, 0);
    argmax_vocab<<<BSZ, 256, 0, stream>>>(out_logits, out_preds, t);
  }
}
